// MultiHeadSelfAttention_13400297963766
// MI455X (gfx1250) — matrix operand order repaired and checked by reading, not compiled, not run on hardware
//
#include <hip/hip_runtime.h>
#include <hip/hip_bf16.h>

typedef __attribute__((ext_vector_type(16))) __bf16 bf16x16;
typedef __attribute__((ext_vector_type(8)))  __bf16 bf16x8;
typedef __attribute__((ext_vector_type(8)))  float  f32x8;

#define D_MODEL   1024
#define NUM_HEADS 16
#define HEAD_DIM  64
#define BATCH     4
#define SEQ       2048
#define MTOT      (BATCH * SEQ)   // 8192 rows

#define LDS_PITCH 40              // 80B row pitch: bank-conflict-free, 16B aligned

union Frag {
  bf16x16 v;
  bf16x8  h[2];
};

__device__ __forceinline__ f32x8 wmma_bf16f32(bf16x16 a, bf16x16 b, f32x8 c) {
  return __builtin_amdgcn_wmma_f32_16x16x32_bf16(false, a, false, b, (short)0, c,
                                                 false, false);
}

// ---- CDNA5 async copy: global -> LDS, 16 bytes per lane, tracked by ASYNCcnt
__device__ __forceinline__ void async_ld_b128(const __bf16* lds_dst,
                                              const __bf16* gsrc) {
  const unsigned lds_off = (unsigned)(unsigned long long)lds_dst;
  asm volatile("global_load_async_to_lds_b128 %0, %1, off"
               :: "v"(lds_off), "v"(gsrc) : "memory");
}

// A fragment (16x32 bf16) from a row-major global matrix (ISA 7.12.2 layout).
__device__ __forceinline__ bf16x16 load_a_frag(const __bf16* __restrict__ base, int ld,
                                               int row0, int k0, int lane) {
  const int r  = row0 + (lane & 15);
  const int kb = k0 + ((lane >> 4) << 3);
  const __bf16* p = base + (long)r * ld + kb;
  Frag f;
  f.h[0] = *(const bf16x8*)(p);
  f.h[1] = *(const bf16x8*)(p + 16);
  return f.v;
}

// B fragment (32x16 bf16, KxN): element (k, n) at base[n*ld + k] (global).
__device__ __forceinline__ bf16x16 load_b_frag(const __bf16* __restrict__ base, long ld,
                                               int n0, int k0, int lane) {
  const int n  = n0 + (lane & 15);
  const int kb = k0 + ((lane >> 4) << 3);
  const __bf16* p = base + (long)n * ld + kb;
  Frag f;
  f.h[0] = *(const bf16x8*)(p);
  f.h[1] = *(const bf16x8*)(p + 16);
  return f.v;
}

// Fragment readers from the staged LDS tiles (row pitch LDS_PITCH elements).
__device__ __forceinline__ bf16x16 lds_a_frag(const __bf16* tile, int row0, int lane) {
  const int r  = row0 + (lane & 15);
  const int kb = (lane >> 4) << 3;
  const __bf16* p = tile + r * LDS_PITCH + kb;
  Frag f;
  f.h[0] = *(const bf16x8*)(p);
  f.h[1] = *(const bf16x8*)(p + 16);
  return f.v;
}

__device__ __forceinline__ bf16x16 lds_b_frag(const __bf16* tile, int n0, int lane) {
  const int n  = n0 + (lane & 15);
  const int kb = (lane >> 4) << 3;
  const __bf16* p = tile + n * LDS_PITCH + kb;
  Frag f;
  f.h[0] = *(const bf16x8*)(p);
  f.h[1] = *(const bf16x8*)(p + 16);
  return f.v;
}

// Stage one 128x32 A tile and one 128x32 B tile into LDS (async).
// 512 16B chunks per tile, 256 threads -> 2 chunks each -> 4 async ops/thread.
__device__ __forceinline__ void stage_tiles(const __bf16* __restrict__ gA,
                                            const __bf16* __restrict__ gB,
                                            __bf16* sA, __bf16* sB,
                                            int k0, int tid) {
#pragma unroll
  for (int i = 0; i < 2; ++i) {
    const int c   = tid + i * 256;
    const int row = c >> 2;
    const int kc  = (c & 3) << 3;
    async_ld_b128(sA + row * LDS_PITCH + kc, gA + (long)row * D_MODEL + k0 + kc);
    async_ld_b128(sB + row * LDS_PITCH + kc, gB + (long)row * D_MODEL + k0 + kc);
  }
}

// ---------------------------------------------------------------- conversion
__global__ __launch_bounds__(256) void cvt_f32_bf16_kernel(
    const float* __restrict__ src, __bf16* __restrict__ dst) {
  const long i = ((long)blockIdx.x * blockDim.x + threadIdx.x) * 8;
  f32x8 v = *(const f32x8*)(src + i);
  bf16x8 o;
#pragma unroll
  for (int j = 0; j < 8; ++j) o[j] = (__bf16)v[j];
  *(bf16x8*)(dst + i) = o;
}

// ---------------------------------------------------------------- QKV GEMM
// C[m,n] = sum_k X[m,k] * W[n,k]. 8 waves (4x2), wave tile 32x64, block 128x128.
// Double-buffered async-to-LDS staging of both operand tiles.
__global__ __launch_bounds__(256) void qkv_gemm_kernel(
    const __bf16* __restrict__ xb,
    const __bf16* __restrict__ wq, const __bf16* __restrict__ wk,
    const __bf16* __restrict__ wv,
    __bf16* __restrict__ Qb, __bf16* __restrict__ Kb, __bf16* __restrict__ Vt) {
  __shared__ __align__(16) __bf16 sA[2][128 * LDS_PITCH];
  __shared__ __align__(16) __bf16 sB[2][128 * LDS_PITCH];

  const int z     = blockIdx.z;
  const __bf16* W = (z == 0) ? wq : (z == 1) ? wk : wv;
  const int tid   = threadIdx.x;
  const int lane  = tid & 31;
  const int wid   = tid >> 5;
  const int m0    = blockIdx.x * 128 + (wid >> 1) * 32;
  const int n0    = blockIdx.y * 128 + (wid & 1) * 64;
  const int m0l   = (wid >> 1) * 32;
  const int n0l   = (wid & 1) * 64;

  const __bf16* gA = xb + (long)blockIdx.x * 128 * D_MODEL;
  const __bf16* gB = W  + (long)blockIdx.y * 128 * D_MODEL;

  stage_tiles(gA, gB, sA[0], sB[0], 0,  tid);
  stage_tiles(gA, gB, sA[1], sB[1], 32, tid);

  f32x8 acc[2][4] = {};
  for (int k0 = 0; k0 < D_MODEL; k0 += 32) {
    const int buf = (k0 >> 5) & 1;
    // In-order ASYNCcnt: <=4 retires everything older than the newest stage.
    if (k0 + 32 < D_MODEL) asm volatile("s_wait_asynccnt 0x4" ::: "memory");
    else                   asm volatile("s_wait_asynccnt 0x0" ::: "memory");
    __syncthreads();   // whole tile (all waves' async ops) now visible

    bf16x16 a0 = lds_a_frag(sA[buf], m0l,      lane);
    bf16x16 a1 = lds_a_frag(sA[buf], m0l + 16, lane);
    bf16x16 bb[4];
#pragma unroll
    for (int nt = 0; nt < 4; ++nt) bb[nt] = lds_b_frag(sB[buf], n0l + nt * 16, lane);
#pragma unroll
    for (int nt = 0; nt < 4; ++nt) {
      acc[0][nt] = wmma_bf16f32(a0, bb[nt], acc[0][nt]);
      acc[1][nt] = wmma_bf16f32(a1, bb[nt], acc[1][nt]);
    }

    __syncthreads();   // everyone done reading buf -> safe to refill it
    if (k0 + 64 < D_MODEL) stage_tiles(gA, gB, sA[buf], sB[buf], k0 + 64, tid);
  }

  const int ncol  = lane & 15;
  const int rbase = (lane >> 4) << 3;
  if (z < 2) {
    __bf16* dst = z ? Kb : Qb;
#pragma unroll
    for (int mt = 0; mt < 2; ++mt)
#pragma unroll
      for (int nt = 0; nt < 4; ++nt)
#pragma unroll
        for (int j = 0; j < 8; ++j) {
          const int m = m0 + mt * 16 + rbase + j;
          const int n = n0 + nt * 16 + ncol;
          dst[(long)m * D_MODEL + n] = (__bf16)acc[mt][nt][j];
        }
  } else {
#pragma unroll
    for (int mt = 0; mt < 2; ++mt)
#pragma unroll
      for (int nt = 0; nt < 4; ++nt) {
        bf16x8 vv;
#pragma unroll
        for (int j = 0; j < 8; ++j) vv[j] = (__bf16)acc[mt][nt][j];
        const int m = m0 + mt * 16 + rbase;   // 8 consecutive rows -> 16B store
        const int n = n0 + nt * 16 + ncol;
        *(bf16x8*)(Vt + (long)n * MTOT + m) = vv;   // V stored transposed
      }
  }
}

// ---------------------------------------------------------------- flash attn
// One wave owns 16 query rows; 8 waves/block -> 128 rows/block. No inter-wave
// barriers (causal loop trip counts diverge per wave).
__global__ __launch_bounds__(256) void flash_attn_kernel(
    const __bf16* __restrict__ Qb, const __bf16* __restrict__ Kb,
    const __bf16* __restrict__ Vt, __bf16* __restrict__ Ab) {
  __shared__ __align__(32) __bf16 plds[8][16 * 32];  // per-wave P transpose tile

  const int lane = threadIdx.x & 31;
  const int wid  = threadIdx.x >> 5;

  const int qblk = blockIdx.x % (SEQ / 128);
  const int bh   = blockIdx.x / (SEQ / 128);
  const int h    = bh % NUM_HEADS;
  const int b    = bh / NUM_HEADS;
  const int q0   = qblk * 128 + wid * 16;

  const __bf16* Qbase = Qb + (long)b * SEQ * D_MODEL + h * HEAD_DIM;
  const __bf16* Kbase = Kb + (long)b * SEQ * D_MODEL + h * HEAD_DIM;
  const __bf16* Vbase = Vt + (long)(h * HEAD_DIM) * MTOT + (long)b * SEQ;

  const bf16x16 qa0 = load_a_frag(Qbase, D_MODEL, q0, 0,  lane);
  const bf16x16 qa1 = load_a_frag(Qbase, D_MODEL, q0, 32, lane);

  f32x8 o[4] = {};          // 16 x 64 output accumulator (4 N-tiles)
  float mrow[8], lrow[8];
#pragma unroll
  for (int j = 0; j < 8; ++j) { mrow[j] = -3.0e38f; lrow[j] = 0.f; }

  const int   ncol  = lane & 15;
  const int   rbase = (lane >> 4) << 3;
  const float sc    = 0.125f * 1.44269504088896f;  // 1/sqrt(64) * log2(e)

  for (int k0 = 0; k0 <= q0 + 15; k0 += 32) {
    // prefetch next K rows / V columns (gfx1250 global_prefetch_b8)
    if (k0 + 32 <= q0 + 15) {
      __builtin_prefetch(Kbase + (long)(k0 + 32 + lane) * D_MODEL, 0, 1);
      __builtin_prefetch(Vbase + (long)(lane << 1) * MTOT + k0 + 32, 0, 1);
    }

    // ---- scores S = Q K^T for columns [k0, k0+32) : two 16-col tiles
    float sv[2][8];
#pragma unroll
    for (int t = 0; t < 2; ++t) {
      f32x8 accs = {};
      bf16x16 kb0 = load_b_frag(Kbase, D_MODEL, k0 + t * 16, 0,  lane);
      accs = wmma_bf16f32(qa0, kb0, accs);
      bf16x16 kb1 = load_b_frag(Kbase, D_MODEL, k0 + t * 16, 32, lane);
      accs = wmma_bf16f32(qa1, kb1, accs);
#pragma unroll
      for (int j = 0; j < 8; ++j) {
        const int kc = k0 + t * 16 + ncol;
        const int qr = q0 + rbase + j;
        const float v = accs[j] * sc;
        sv[t][j] = (kc <= qr) ? v : -3.0e38f;   // causal mask
      }
    }

    // ---- online softmax: row reductions across 16-lane halves (wave32)
#pragma unroll
    for (int j = 0; j < 8; ++j) {
      float mj = fmaxf(sv[0][j], sv[1][j]);
#pragma unroll
      for (int off = 1; off < 16; off <<= 1)
        mj = fmaxf(mj, __shfl_xor(mj, off, 32));
      const float mn    = fmaxf(mrow[j], mj);
      const float alpha = __builtin_amdgcn_exp2f(mrow[j] - mn);
      mrow[j] = mn;
      lrow[j] *= alpha;
#pragma unroll
      for (int nt = 0; nt < 4; ++nt) o[nt][j] *= alpha;
      float psum = 0.f;
#pragma unroll
      for (int t = 0; t < 2; ++t) {
        const float p = __builtin_amdgcn_exp2f(sv[t][j] - mn);
        psum += p;
        plds[wid][(rbase + j) * 32 + t * 16 + ncol] = (__bf16)p;
      }
#pragma unroll
      for (int off = 1; off < 16; off <<= 1)
        psum += __shfl_xor(psum, off, 32);
      lrow[j] += psum;
    }

    // ---- P (C-layout) -> A-layout via per-wave LDS tile (DS ops from one
    // wave are in-order; dscnt wait covers the cross-lane visibility).
    asm volatile("s_wait_dscnt 0x0" ::: "memory");
    Frag pa;
    {
      const int prow = lane & 15;
      const int pkb  = (lane >> 4) << 3;
      pa.h[0] = *(const bf16x8*)&plds[wid][prow * 32 + pkb];
      pa.h[1] = *(const bf16x8*)&plds[wid][prow * 32 + pkb + 16];
    }

    // ---- O += P * V  (V pre-transposed: contiguous along K)
#pragma unroll
    for (int nt = 0; nt < 4; ++nt) {
      bf16x16 vb = load_b_frag(Vbase, MTOT, nt * 16, k0, lane);
      o[nt] = wmma_bf16f32(pa.v, vb, o[nt]);
    }
  }

  float inv[8];
#pragma unroll
  for (int j = 0; j < 8; ++j) inv[j] = __builtin_amdgcn_rcpf(lrow[j]);
#pragma unroll
  for (int nt = 0; nt < 4; ++nt)
#pragma unroll
    for (int j = 0; j < 8; ++j) {
      const int qr = q0 + rbase + j;
      const int dc = h * HEAD_DIM + nt * 16 + ncol;
      Ab[(long)(b * SEQ + qr) * D_MODEL + dc] = (__bf16)(o[nt][j] * inv[j]);
    }
}

// ---------------------------------------------------------------- out proj
__global__ __launch_bounds__(256) void out_proj_kernel(
    const __bf16* __restrict__ Ab, const __bf16* __restrict__ wo,
    float* __restrict__ out) {
  __shared__ __align__(16) __bf16 sA[2][128 * LDS_PITCH];
  __shared__ __align__(16) __bf16 sB[2][128 * LDS_PITCH];

  const int tid  = threadIdx.x;
  const int lane = tid & 31;
  const int wid  = tid >> 5;
  const int m0   = blockIdx.x * 128 + (wid >> 1) * 32;
  const int n0   = blockIdx.y * 128 + (wid & 1) * 64;
  const int m0l  = (wid >> 1) * 32;
  const int n0l  = (wid & 1) * 64;

  const __bf16* gA = Ab + (long)blockIdx.x * 128 * D_MODEL;
  const __bf16* gB = wo + (long)blockIdx.y * 128 * D_MODEL;

  stage_tiles(gA, gB, sA[0], sB[0], 0,  tid);
  stage_tiles(gA, gB, sA[1], sB[1], 32, tid);

  f32x8 acc[2][4] = {};
  for (int k0 = 0; k0 < D_MODEL; k0 += 32) {
    const int buf = (k0 >> 5) & 1;
    if (k0 + 32 < D_MODEL) asm volatile("s_wait_asynccnt 0x4" ::: "memory");
    else                   asm volatile("s_wait_asynccnt 0x0" ::: "memory");
    __syncthreads();

    bf16x16 a0 = lds_a_frag(sA[buf], m0l,      lane);
    bf16x16 a1 = lds_a_frag(sA[buf], m0l + 16, lane);
    bf16x16 bb[4];
#pragma unroll
    for (int nt = 0; nt < 4; ++nt) bb[nt] = lds_b_frag(sB[buf], n0l + nt * 16, lane);
#pragma unroll
    for (int nt = 0; nt < 4; ++nt) {
      acc[0][nt] = wmma_bf16f32(a0, bb[nt], acc[0][nt]);
      acc[1][nt] = wmma_bf16f32(a1, bb[nt], acc[1][nt]);
    }

    __syncthreads();
    if (k0 + 64 < D_MODEL) stage_tiles(gA, gB, sA[buf], sB[buf], k0 + 64, tid);
  }

  const int ncol  = lane & 15;
  const int rbase = (lane >> 4) << 3;
#pragma unroll
  for (int mt = 0; mt < 2; ++mt)
#pragma unroll
    for (int nt = 0; nt < 4; ++nt)
#pragma unroll
      for (int j = 0; j < 8; ++j) {
        const int m = m0 + mt * 16 + rbase + j;
        const int n = n0 + nt * 16 + ncol;
        out[(long)m * D_MODEL + n] = acc[mt][nt][j];
      }
}

// ---------------------------------------------------------------- launch
extern "C" void kernel_launch(void* const* d_in, const int* in_sizes, int n_in,
                              void* d_out, int out_size, void* d_ws, size_t ws_size,
                              hipStream_t stream) {
  const float* x  = (const float*)d_in[0];
  const float* Wq = (const float*)d_in[1];
  const float* Wk = (const float*)d_in[2];
  const float* Wv = (const float*)d_in[3];
  const float* Wo = (const float*)d_in[4];

  char* ws = (char*)d_ws;
  const size_t XB = (size_t)MTOT * D_MODEL * sizeof(__bf16);     // 16 MiB
  const size_t WB = (size_t)D_MODEL * D_MODEL * sizeof(__bf16);  //  2 MiB
  __bf16* xb  = (__bf16*)(ws);
  __bf16* wqb = (__bf16*)(ws + XB);
  __bf16* wkb = (__bf16*)(ws + XB + 1 * WB);
  __bf16* wvb = (__bf16*)(ws + XB + 2 * WB);
  __bf16* wob = (__bf16*)(ws + XB + 3 * WB);
  __bf16* Qb  = (__bf16*)(ws + 1 * XB + 4 * WB);
  __bf16* Kb  = (__bf16*)(ws + 2 * XB + 4 * WB);
  __bf16* Vt  = (__bf16*)(ws + 3 * XB + 4 * WB);
  __bf16* Ab  = (__bf16*)(ws + 4 * XB + 4 * WB);  // total 88 MiB

  const int NX = MTOT * D_MODEL;      // 8388608
  const int NW = D_MODEL * D_MODEL;   // 1048576
  cvt_f32_bf16_kernel<<<NX / 2048, 256, 0, stream>>>(x,  xb);
  cvt_f32_bf16_kernel<<<NW / 2048, 256, 0, stream>>>(Wq, wqb);
  cvt_f32_bf16_kernel<<<NW / 2048, 256, 0, stream>>>(Wk, wkb);
  cvt_f32_bf16_kernel<<<NW / 2048, 256, 0, stream>>>(Wv, wvb);
  cvt_f32_bf16_kernel<<<NW / 2048, 256, 0, stream>>>(Wo, wob);

  qkv_gemm_kernel<<<dim3(MTOT / 128, D_MODEL / 128, 3), 256, 0, stream>>>(
      xb, wqb, wkb, wvb, Qb, Kb, Vt);

  flash_attn_kernel<<<dim3(BATCH * NUM_HEADS * (SEQ / 128)), 256, 0, stream>>>(
      Qb, Kb, Vt, Ab);

  out_proj_kernel<<<dim3(MTOT / 128, D_MODEL / 128), 256, 0, stream>>>(
      Ab, wob, (float*)d_out);
}
